// SEAN_Generator_70334384439493
// MI455X (gfx1250) — matrix operand order repaired and checked by reading, not compiled, not run on hardware
//
#include <hip/hip_runtime.h>
#include <hip/hip_bf16.h>
#include <math.h>
#include <stdint.h>

typedef __bf16 bf16;
typedef __attribute__((ext_vector_type(16))) __bf16 v16bf;
typedef __attribute__((ext_vector_type(8)))  float  v8f;

#define EPS 1e-5f
// activation codes
#define ACT_NONE  0
#define ACT_LRELU 1
#define ACT_RELU  2
#define ACT_TANH  3

// ---------------------------------------------------------------------------
// Elementwise / utility kernels
// ---------------------------------------------------------------------------

__global__ void k_fill_f32(float* p, float v, long n) {
  long i = (long)blockIdx.x * blockDim.x + threadIdx.x;
  if (i < n) p[i] = v;
}

__global__ void k_f32_to_bf16_act(const float* __restrict__ x, bf16* __restrict__ y,
                                  long n, int act) {
  long i = (long)blockIdx.x * blockDim.x + threadIdx.x;
  if (i < n) {
    float v = x[i];
    if (act == ACT_LRELU) v = v > 0.f ? v : 0.2f * v;
    else if (act == ACT_RELU) v = fmaxf(v, 0.f);
    y[i] = (bf16)v;
  }
}

__global__ void k_add(const float* __restrict__ a, const float* __restrict__ b,
                      float* __restrict__ y, long n) {
  long i = (long)blockIdx.x * blockDim.x + threadIdx.x;
  if (i < n) y[i] = a[i] + b[i];
}

// nearest upsample/resize, matches ih = (oh*H)//OH
__global__ void k_upsample(const float* __restrict__ x, int C, int H, int W,
                           float* __restrict__ y, int OH, int OW, long n) {
  long i = (long)blockIdx.x * blockDim.x + threadIdx.x;
  if (i < n) {
    int ow = (int)(i % OW);
    long t = i / OW;
    int oh = (int)(t % OH);
    int c = (int)(t / OH);
    int ih = (oh * H) / OH, iw = (ow * W) / OW;
    y[i] = x[((long)c * H + ih) * W + iw];
  }
}

// reflect pad by 1 (NCHW, C planes)
__global__ void k_reflectpad(const float* __restrict__ x, int C, int H, int W,
                             float* __restrict__ y, long n) {
  long i = (long)blockIdx.x * blockDim.x + threadIdx.x;
  if (i < n) {
    int OW = W + 2, OH = H + 2;
    int ow = (int)(i % OW);
    long t = i / OW;
    int oh = (int)(t % OH);
    int c = (int)(t / OH);
    int ih = oh - 1, iw = ow - 1;
    if (ih < 0) ih = -ih;
    if (ih >= H) ih = 2 * H - 2 - ih;
    if (iw < 0) iw = -iw;
    if (iw >= W) iw = 2 * W - 2 - iw;
    y[i] = x[((long)c * H + ih) * W + iw];
  }
}

// zero-stuffed dilation for ConvTranspose(k3,s2,p1,op1):
// y (C, 2H+2, 2W+2) prezeroed; y[c][1+2h][1+2w] = x[c][h][w]
__global__ void k_dilate2(const float* __restrict__ x, float* __restrict__ y,
                          int C, int H, int W, long n) {
  long i = (long)blockIdx.x * blockDim.x + threadIdx.x;
  if (i < n) {
    int w = (int)(i % W);
    long t = i / W;
    int h = (int)(t % H);
    int c = (int)(t / H);
    int DH = 2 * H + 2, DW = 2 * W + 2;
    y[((long)c * DH + (1 + 2 * h)) * DW + (1 + 2 * w)] = x[i];
  }
}

// pack f32 OIHW weights -> zero-padded bf16 [Coutp][Kp]
__global__ void k_wpack(const float* __restrict__ w, bf16* __restrict__ wp,
                        int Cout, int K, int Kp, long n) {
  long i = (long)blockIdx.x * blockDim.x + threadIdx.x;
  if (i < n) {
    int kg = (int)(i % Kp);
    int o = (int)(i / Kp);
    float v = 0.f;
    if (o < Cout && kg < K) v = w[(long)o * K + kg];
    wp[i] = (bf16)v;
  }
}

// ConvTranspose weights (Cin,Cout,3,3) -> padded bf16 [Coutp][Kp], OIHW flipped
__global__ void k_wtflip(const float* __restrict__ w, bf16* __restrict__ wp,
                         int Cin, int Cout, int Kp, long n) {
  long i = (long)blockIdx.x * blockDim.x + threadIdx.x;
  if (i < n) {
    int kg = (int)(i % Kp);
    int o = (int)(i / Kp);
    float v = 0.f;
    if (o < Cout && kg < Cin * 9) {
      int ci = kg / 9;
      int r = kg - ci * 9;
      int kh = r / 3, kw = r - kh * 3;
      v = w[(((long)ci * Cout + o) * 3 + (2 - kh)) * 3 + (2 - kw)];
    }
    wp[i] = (bf16)v;
  }
}

// per-channel mean / inv-std (B=1 -> batch_norm == instance_norm)
__global__ void k_meanvar(const float* __restrict__ x, float* mean, float* istd, int HW) {
  int c = blockIdx.x;
  const float* p = x + (long)c * HW;
  float s = 0.f, s2 = 0.f;
  for (int i = threadIdx.x; i < HW; i += blockDim.x) {
    float v = p[i];
    s += v; s2 += v * v;
  }
  __shared__ float sh0[256];
  __shared__ float sh1[256];
  sh0[threadIdx.x] = s; sh1[threadIdx.x] = s2;
  __syncthreads();
  for (int o = 128; o > 0; o >>= 1) {
    if (threadIdx.x < o) { sh0[threadIdx.x] += sh0[threadIdx.x + o]; sh1[threadIdx.x] += sh1[threadIdx.x + o]; }
    __syncthreads();
  }
  if (threadIdx.x == 0) {
    float m = sh0[0] / (float)HW;
    float v = sh1[0] / (float)HW - m * m;
    mean[c] = m;
    istd[c] = rsqrtf(v + EPS);
  }
}

__global__ void k_norm_act(const float* __restrict__ x, const float* __restrict__ mean,
                           const float* __restrict__ istd, float* __restrict__ y,
                           int HW, long n, int act) {
  long i = (long)blockIdx.x * blockDim.x + threadIdx.x;
  if (i < n) {
    int c = (int)(i / HW);
    float v = (x[i] - mean[c]) * istd[c];
    if (act == ACT_LRELU) v = v > 0.f ? v : 0.2f * v;
    y[i] = v;
  }
}

// SPADE modulation: y = xn*(1+gamma)+beta, gamma/beta blended by sigmoid(bg/bb)
__global__ void k_spade(const float* __restrict__ xn,
                        const float* __restrict__ gsp, const float* __restrict__ bsp,
                        const float* __restrict__ gst, const float* __restrict__ bst,
                        const float* __restrict__ bg, const float* __restrict__ bb,
                        float* __restrict__ y, long n, int act, int useST) {
  long i = (long)blockIdx.x * blockDim.x + threadIdx.x;
  if (i >= n) return;
  float g, b;
  if (useST) {
    float ga = 1.f / (1.f + expf(-bg[0]));
    float ba = 1.f / (1.f + expf(-bb[0]));
    g = ga * gst[i] + (1.f - ga) * gsp[i];
    b = ba * bst[i] + (1.f - ba) * bsp[i];
  } else {
    g = gsp[i];
    b = bsp[i];
  }
  float v = xn[i] * (1.f + g) + b;
  if (act == ACT_LRELU) v = v > 0.f ? v : 0.2f * v;
  y[i] = v;
}

// label map (argmax of one-hot seg at 256x256)
__global__ void k_labels(const float* __restrict__ seg, int* __restrict__ lab, int P) {
  int p = blockIdx.x * blockDim.x + threadIdx.x;
  if (p < P) {
    int best = 0;
    float bv = seg[p];
    for (int j = 1; j < 19; ++j) {
      float v = seg[(long)j * P + p];
      if (v > bv) { bv = v; best = j; }
    }
    lab[p] = best;
  }
}

// one-hot segmap at SxS (bf16, conv-ready), sampled nearest from 256x256 labels
__global__ void k_segmap(const int* __restrict__ lab256, bf16* __restrict__ out,
                         int S, long n) {
  long i = (long)blockIdx.x * blockDim.x + threadIdx.x;
  if (i < n) {
    long P = (long)S * S;
    int j = (int)(i / P);
    long p = i - (long)j * P;
    int h = (int)(p / S), w = (int)(p % S);
    int lab = lab256[((h * 256) / S) * 256 + ((w * 256) / S)];
    out[i] = (bf16)((lab == j) ? 1.f : 0.f);
  }
}

// style map: out[c][h][w] = middle[label(h,w)][c]
__global__ void k_style_map(const int* __restrict__ lab256, const float* __restrict__ mid,
                            bf16* __restrict__ out, int S, long n) {
  long i = (long)blockIdx.x * blockDim.x + threadIdx.x;
  if (i < n) {
    long P = (long)S * S;
    int c = (int)(i / P);
    long p = i - (long)c * P;
    int h = (int)(p / S), w = (int)(p % S);
    int lab = lab256[((h * 256) / S) * 256 + ((w * 256) / S)];
    out[i] = (bf16)mid[lab * 512 + c];
  }
}

// per-class pixel counts at 128x128 (area of seg128)
__global__ void k_area(const int* __restrict__ lab256, float* __restrict__ area, int P) {
  int p = blockIdx.x * blockDim.x + threadIdx.x;
  if (p < P) {
    int h = p / 128, w = p % 128;
    atomicAdd(&area[lab256[(2 * h) * 256 + 2 * w]], 1.f);
  }
}

// sums[j][c] += u[c][p] for pixels of class j (u: 512 x 128x128)
__global__ void k_class_sum(const float* __restrict__ u, const int* __restrict__ lab256,
                            float* __restrict__ sums, int P, long n) {
  long i = (long)blockIdx.x * blockDim.x + threadIdx.x;
  if (i < n) {
    int c = (int)(i / P);
    int p = (int)(i % P);
    int h = p / 128, w = p % 128;
    int lab = lab256[(2 * h) * 256 + 2 * w];
    atomicAdd(&sums[lab * 512 + c], u[i]);
  }
}

__global__ void k_style_final(const float* __restrict__ sums, const float* __restrict__ area,
                              float* __restrict__ style, int n) {
  int i = blockIdx.x * blockDim.x + threadIdx.x;
  if (i < n) {
    int j = i / 512;
    float a = area[j];
    style[i] = a > 0.f ? sums[i] / fmaxf(a, 1.f) : 0.f;
  }
}

// per-class Linear(512,512)+ReLU
__global__ void k_style_mlp(const float* __restrict__ style, const float* __restrict__ stw,
                            const float* __restrict__ stb, float* __restrict__ mid) {
  int j = blockIdx.x;
  __shared__ float sh[512];
  for (int d = threadIdx.x; d < 512; d += blockDim.x) sh[d] = style[j * 512 + d];
  __syncthreads();
  for (int o = threadIdx.x; o < 512; o += blockDim.x) {
    const float* wr = stw + ((long)j * 512 + o) * 512;
    float acc = stb[j * 512 + o];
    for (int d = 0; d < 512; ++d) acc += sh[d] * wr[d];
    mid[j * 512 + o] = fmaxf(acc, 0.f);
  }
}

// ---------------------------------------------------------------------------
// Implicit-GEMM convolution via V_WMMA_F32_16X16X32_BF16
//   block = 128 threads = 4 wave32 waves; block tile = 32 pixels x 64 out-ch
//   each wave: 16 pixels x 32 channels = 2 WMMA tiles sharing one A fragment
//   B (weight) panel staged with GLOBAL_LOAD_ASYNC_TO_LDS_B128 (ASYNCcnt),
//   weights pre-packed zero-padded [ceil64(Cout)][ceil32(K)] so the async
//   path is unconditional and 16B aligned.
// ---------------------------------------------------------------------------
#define TM 32
#define TN 64
#define TK 32
#define LDA 40  // bf16 elems per As row (80B, 16B-aligned, bank-spread)
#define LDB 40  // bf16 elems per Bs row

__launch_bounds__(128)
__global__ void k_conv_wmma(const bf16* __restrict__ X, const bf16* __restrict__ Wt,
                            const float* __restrict__ bias, float* __restrict__ Y,
                            int Cin, int H, int W, int Cout, int OH, int OW,
                            int KH, int KW, int stride, int pad, int Kp, int act) {
  __shared__ bf16 As[TM][LDA];
  __shared__ bf16 Bs[TN][LDB];

  const int K = Cin * KH * KW;
  const int P = OH * OW;
  const int p0 = blockIdx.x * TM;
  const int n0 = blockIdx.y * TN;
  const int tid = (int)threadIdx.x;
  const int lane = tid & 31;
  const int wave = tid >> 5;
  const int wm = (wave & 1) * 16;   // pixel sub-tile
  const int wn = (wave >> 1) * 32;  // 32-channel sub-tile (2 WMMA tiles)

  // async B staging coordinates: 64 rows x 64B = 256 x 16B chunks, 2/thread
  const int bn = tid >> 2;          // 0..31
  const int bp = (tid & 3) * 8;     // bf16 offset within row (0,8,16,24)
  const unsigned lds0 = (unsigned)(uintptr_t)&Bs[bn][bp];
  const unsigned lds1 = (unsigned)(uintptr_t)&Bs[bn + 32][bp];

  v8f acc0 = {0.f, 0.f, 0.f, 0.f, 0.f, 0.f, 0.f, 0.f};
  v8f acc1 = {0.f, 0.f, 0.f, 0.f, 0.f, 0.f, 0.f, 0.f};

  for (int kk = 0; kk < K; kk += TK) {
    // ---- async DMA: weight panel global -> LDS (no VGPR round-trip) ----
    {
      unsigned g0 = ((unsigned)(n0 + bn) * (unsigned)Kp + (unsigned)(kk) + bp) * 2u;
      unsigned g1 = ((unsigned)(n0 + bn + 32) * (unsigned)Kp + (unsigned)(kk) + bp) * 2u;
      asm volatile("global_load_async_to_lds_b128 %0, %1, %2"
                   :: "v"(lds0), "v"(g0), "s"(Wt) : "memory");
      asm volatile("global_load_async_to_lds_b128 %0, %1, %2"
                   :: "v"(lds1), "v"(g1), "s"(Wt) : "memory");
    }
    // ---- manual stage A (im2col with bounds + zero fill) ----
    #pragma unroll 2
    for (int t = tid; t < TM * TK; t += 128) {
      int pl = t / TK, kl = t % TK;
      int p = p0 + pl, kg = kk + kl;
      float v = 0.f;
      if (p < P && kg < K) {
        int khw = KH * KW;
        int ci = kg / khw;
        int r = kg - ci * khw;
        int kh = r / KW, kw = r - kh * KW;
        int oh = p / OW, ow = p - oh * OW;
        int ih = oh * stride - pad + kh;
        int iw = ow * stride - pad + kw;
        if ((unsigned)ih < (unsigned)H && (unsigned)iw < (unsigned)W)
          v = (float)X[(long)ci * H * W + (long)ih * W + iw];
      }
      As[pl][kl] = (bf16)v;
    }
    asm volatile("s_wait_asynccnt 0x0" ::: "memory");
    __syncthreads();

    // ---- pack fragments per CDNA5 WMMA VGPR layouts (ISA 7.12.2) ----
    v16bf a, b0, b1;
    {
      const int m = lane & 15;
      const int hi = lane >> 4;
      #pragma unroll
      for (int v2 = 0; v2 < 8; ++v2) {
        int kb = ((v2 < 4) ? 0 : 16) + hi * 8 + (v2 & 3) * 2;
        a[2 * v2] = As[wm + m][kb];
        a[2 * v2 + 1] = As[wm + m][kb + 1];
      }
      // B fragment: lane holds column n = lane&15, K split by lane half;
      // Bs is [N][K] row-major -> same K order as the A fragment
      #pragma unroll
      for (int v2 = 0; v2 < 8; ++v2) {
        int kb = ((v2 < 4) ? 0 : 16) + hi * 8 + (v2 & 3) * 2;
        b0[2 * v2] = Bs[wn + m][kb];
        b0[2 * v2 + 1] = Bs[wn + m][kb + 1];
        b1[2 * v2] = Bs[wn + 16 + m][kb];
        b1[2 * v2 + 1] = Bs[wn + 16 + m][kb + 1];
      }
    }
    acc0 = __builtin_amdgcn_wmma_f32_16x16x32_bf16(false, a, false, b0,
                                                   (short)0, acc0, false, false);
    acc1 = __builtin_amdgcn_wmma_f32_16x16x32_bf16(false, a, false, b1,
                                                   (short)0, acc1, false, false);
    __syncthreads();
  }

  // ---- epilogue: D layout M = r + 8*(lane>=16), N = lane&15 ----
  const int n = lane & 15;
  const int hi = lane >> 4;
  #pragma unroll
  for (int half = 0; half < 2; ++half) {
    const int co = n0 + wn + half * 16 + n;
    if (co < Cout) {
      float bs = bias ? bias[co] : 0.f;
      #pragma unroll
      for (int r = 0; r < 8; ++r) {
        int p = p0 + wm + r + 8 * hi;
        if (p < P) {
          float v = (half ? acc1[r] : acc0[r]) + bs;
          if (act == ACT_LRELU) v = v > 0.f ? v : 0.2f * v;
          else if (act == ACT_RELU) v = fmaxf(v, 0.f);
          else if (act == ACT_TANH) v = tanhf(v);
          Y[(long)co * P + p] = v;
        }
      }
    }
  }
}

// ---------------------------------------------------------------------------
// Host-side orchestration
// ---------------------------------------------------------------------------

namespace {

struct Arena {
  char* base;
  size_t off, cap;
  void* take(size_t bytes) {
    size_t o = (off + 255) & ~(size_t)255;
    off = o + bytes;
    return base + o;
  }
};

inline dim3 gs(long n) { return dim3((unsigned)((n + 255) / 256)); }
inline int cdiv(int a, int b) { return (a + b - 1) / b; }

struct DetailP {
  const float *noise_var, *bg, *bb, *sp_w, *sp_b, *g_w, *g_b, *b_w, *b_b;
  const float *st_w, *st_b, *cg_w, *cg_b, *cb_w, *cb_b;
};
struct ResP {
  DetailP d0, d1, ds;
  const float *c0_w, *c0_b, *c1_w, *c1_b, *cs_w;
};
struct EncP {
  const float *c1w, *c1b, *c2w, *c2b, *c3w, *c3b, *ctw, *ctb, *c4w, *c4b;
};

void conv_core(hipStream_t st, const bf16* X, int Cin, int H, int W,
               const bf16* Wb, const float* Bf, int Cout, int KH, int KW,
               int stride, int pad, int Kp, float* Y, int OH, int OW, int act) {
  dim3 g(cdiv(OH * OW, TM), cdiv(Cout, TN));
  k_conv_wmma<<<g, 128, 0, st>>>(X, Wb, Bf, Y, Cin, H, W, Cout, OH, OW,
                                 KH, KW, stride, pad, Kp, act);
}

void conv_run(hipStream_t st, Arena& A, const bf16* X, int Cin, int H, int W,
              const float* Wf, const float* Bf, int Cout, int KH, int KW,
              int stride, int pad, float* Y, int OH, int OW, int act) {
  const int K = Cin * KH * KW;
  const int Kp = cdiv(K, TK) * TK;
  const int Cp = cdiv(Cout, TN) * TN;
  long wn = (long)Cp * Kp;
  bf16* Wb = (bf16*)A.take((size_t)wn * sizeof(bf16));
  k_wpack<<<gs(wn), 256, 0, st>>>(Wf, Wb, Cout, K, Kp, wn);
  conv_core(st, X, Cin, H, W, Wb, Bf, Cout, KH, KW, stride, pad, Kp, Y, OH, OW, act);
}

void norm_inplace(hipStream_t st, Arena& A, float* x, int C, int HW, int act) {
  float* mean = (float*)A.take((size_t)C * 4);
  float* istd = (float*)A.take((size_t)C * 4);
  k_meanvar<<<C, 256, 0, st>>>(x, mean, istd, HW);
  long n = (long)C * HW;
  k_norm_act<<<gs(n), 256, 0, st>>>(x, mean, istd, x, HW, n, act);
}

// SEAN "detail" (SPADE+style) block; returns modulated tensor (act_out applied)
float* detail_blk(hipStream_t st, Arena& A, const DetailP& d, int C, bool useST,
                  const float* x, int S, const int* lab256, const float* style,
                  int act_out) {
  const int P = S * S;
  const long CP = (long)C * P;
  // batch-norm (B=1): per-channel stats; noise path dropped (noise_var == 0)
  float* mean = (float*)A.take((size_t)C * 4);
  float* istd = (float*)A.take((size_t)C * 4);
  k_meanvar<<<C, 256, 0, st>>>(x, mean, istd, P);
  float* xn = (float*)A.take((size_t)CP * 4);
  k_norm_act<<<gs(CP), 256, 0, st>>>(x, mean, istd, xn, P, CP, ACT_NONE);

  // spatial SPADE path: segmap -> shared conv (relu) -> gamma/beta convs
  bf16* seg = (bf16*)A.take((size_t)19 * P * sizeof(bf16));
  k_segmap<<<gs(19L * P), 256, 0, st>>>(lab256, seg, S, 19L * P);
  float* actv = (float*)A.take((size_t)128 * P * 4);
  conv_run(st, A, seg, 19, S, S, d.sp_w, d.sp_b, 128, 3, 3, 1, 1, actv, S, S, ACT_RELU);
  bf16* actv_b = (bf16*)A.take((size_t)128 * P * sizeof(bf16));
  k_f32_to_bf16_act<<<gs(128L * P), 256, 0, st>>>(actv, actv_b, 128L * P, ACT_NONE);
  float* gsp = (float*)A.take((size_t)CP * 4);
  float* bsp = (float*)A.take((size_t)CP * 4);
  conv_run(st, A, actv_b, 128, S, S, d.g_w, d.g_b, C, 3, 3, 1, 1, gsp, S, S, ACT_NONE);
  conv_run(st, A, actv_b, 128, S, S, d.b_w, d.b_b, C, 3, 3, 1, 1, bsp, S, S, ACT_NONE);

  float *gst = nullptr, *bst = nullptr;
  if (useST) {
    float* mid = (float*)A.take((size_t)19 * 512 * 4);
    k_style_mlp<<<19, 256, 0, st>>>(style, d.st_w, d.st_b, mid);
    bf16* smap = (bf16*)A.take((size_t)512 * P * sizeof(bf16));
    k_style_map<<<gs(512L * P), 256, 0, st>>>(lab256, mid, smap, S, 512L * P);
    gst = (float*)A.take((size_t)CP * 4);
    bst = (float*)A.take((size_t)CP * 4);
    conv_run(st, A, smap, 512, S, S, d.cg_w, d.cg_b, C, 3, 3, 1, 1, gst, S, S, ACT_NONE);
    conv_run(st, A, smap, 512, S, S, d.cb_w, d.cb_b, C, 3, 3, 1, 1, bst, S, S, ACT_NONE);
  }
  float* out = (float*)A.take((size_t)CP * 4);
  k_spade<<<gs(CP), 256, 0, st>>>(xn, gsp, bsp, gst, bst, d.bg, d.bb, out, CP,
                                  act_out, useST ? 1 : 0);
  return out;
}

float* resblk_run(hipStream_t st, Arena& A, const ResP& r, int cin, int cout,
                  bool useST, const float* h, int S, const int* lab256,
                  const float* style) {
  const int mid = cin < cout ? cin : cout;
  const int P = S * S;
  float* h0 = (float*)A.take((size_t)mid * P * 4);
  float* h1 = (float*)A.take((size_t)cout * P * 4);
  float* sb = (float*)A.take((size_t)cout * P * 4);
  size_t mk = A.off;

  {  // h0 = conv(lrelu(detail(d0, h)), c0)
    float* t = detail_blk(st, A, r.d0, cin, useST, h, S, lab256, style, ACT_LRELU);
    bf16* tb = (bf16*)A.take((size_t)cin * P * sizeof(bf16));
    k_f32_to_bf16_act<<<gs((long)cin * P), 256, 0, st>>>(t, tb, (long)cin * P, ACT_NONE);
    conv_run(st, A, tb, cin, S, S, r.c0_w, r.c0_b, mid, 3, 3, 1, 1, h0, S, S, ACT_NONE);
  }
  A.off = mk;
  {  // h1 = conv(lrelu(detail(d1, h0)), c1)
    float* t = detail_blk(st, A, r.d1, mid, useST, h0, S, lab256, style, ACT_LRELU);
    bf16* tb = (bf16*)A.take((size_t)mid * P * sizeof(bf16));
    k_f32_to_bf16_act<<<gs((long)mid * P), 256, 0, st>>>(t, tb, (long)mid * P, ACT_NONE);
    conv_run(st, A, tb, mid, S, S, r.c1_w, r.c1_b, cout, 3, 3, 1, 1, h1, S, S, ACT_NONE);
  }
  A.off = mk;
  {  // shortcut: conv1x1(detail(ds, h)), no bias, no activation
    float* t = detail_blk(st, A, r.ds, cin, useST, h, S, lab256, style, ACT_NONE);
    bf16* tb = (bf16*)A.take((size_t)cin * P * sizeof(bf16));
    k_f32_to_bf16_act<<<gs((long)cin * P), 256, 0, st>>>(t, tb, (long)cin * P, ACT_NONE);
    conv_run(st, A, tb, cin, S, S, r.cs_w, nullptr, cout, 1, 1, 1, 0, sb, S, S, ACT_NONE);
  }
  A.off = mk;
  k_add<<<gs((long)cout * P), 256, 0, st>>>(h1, sb, h1, (long)cout * P);
  return h1;
}

void encoder_run(hipStream_t st, Arena& A, const EncP& e, const float* img,
                 const int* lab256, float* style) {
  // reflect-pad + conv c1 (VALID) -> (32,256,256)
  float* p1 = (float*)A.take(3L * 258 * 258 * 4);
  k_reflectpad<<<gs(3L * 258 * 258), 256, 0, st>>>(img, 3, 256, 256, p1, 3L * 258 * 258);
  bf16* xb = (bf16*)A.take(3L * 258 * 258 * sizeof(bf16));
  k_f32_to_bf16_act<<<gs(3L * 258 * 258), 256, 0, st>>>(p1, xb, 3L * 258 * 258, ACT_NONE);
  float* a1 = (float*)A.take(32L * 65536 * 4);
  conv_run(st, A, xb, 3, 258, 258, e.c1w, e.c1b, 32, 3, 3, 1, 0, a1, 256, 256, ACT_NONE);
  norm_inplace(st, A, a1, 32, 65536, ACT_LRELU);

  bf16* b1 = (bf16*)A.take(32L * 65536 * sizeof(bf16));
  k_f32_to_bf16_act<<<gs(32L * 65536), 256, 0, st>>>(a1, b1, 32L * 65536, ACT_NONE);
  float* a2 = (float*)A.take(64L * 16384 * 4);
  conv_run(st, A, b1, 32, 256, 256, e.c2w, e.c2b, 64, 3, 3, 2, 1, a2, 128, 128, ACT_NONE);
  norm_inplace(st, A, a2, 64, 16384, ACT_LRELU);

  bf16* b2 = (bf16*)A.take(64L * 16384 * sizeof(bf16));
  k_f32_to_bf16_act<<<gs(64L * 16384), 256, 0, st>>>(a2, b2, 64L * 16384, ACT_NONE);
  float* a3 = (float*)A.take(128L * 4096 * 4);
  conv_run(st, A, b2, 64, 128, 128, e.c3w, e.c3b, 128, 3, 3, 2, 1, a3, 64, 64, ACT_NONE);
  norm_inplace(st, A, a3, 128, 4096, ACT_LRELU);

  // ConvTranspose(k3,s2,p1,op1) == zero-dilate + asym pad + VALID conv(flipped W^T)
  long dn = 128L * 130 * 130;
  float* dil = (float*)A.take(dn * 4);
  k_fill_f32<<<gs(dn), 256, 0, st>>>(dil, 0.f, dn);
  k_dilate2<<<gs(128L * 4096), 256, 0, st>>>(a3, dil, 128, 64, 64, 128L * 4096);
  bf16* dilb = (bf16*)A.take(dn * sizeof(bf16));
  k_f32_to_bf16_act<<<gs(dn), 256, 0, st>>>(dil, dilb, dn, ACT_NONE);
  {
    const int K = 128 * 9;               // 1152
    const int Kp = cdiv(K, TK) * TK;     // 1152
    const int Cp = cdiv(256, TN) * TN;   // 256
    long wn = (long)Cp * Kp;
    bf16* wtb = (bf16*)A.take((size_t)wn * sizeof(bf16));
    k_wtflip<<<gs(wn), 256, 0, st>>>(e.ctw, wtb, 128, 256, Kp, wn);
    float* u1 = (float*)A.take(256L * 16384 * 4);
    conv_core(st, dilb, 128, 130, 130, wtb, e.ctb, 256, 3, 3, 1, 0, Kp, u1, 128, 128, ACT_NONE);
    norm_inplace(st, A, u1, 256, 16384, ACT_NONE);

    long pn = 256L * 130 * 130;
    float* p2 = (float*)A.take(pn * 4);
    k_reflectpad<<<gs(pn), 256, 0, st>>>(u1, 256, 128, 128, p2, pn);
    bf16* p2b = (bf16*)A.take(pn * sizeof(bf16));
    k_f32_to_bf16_act<<<gs(pn), 256, 0, st>>>(p2, p2b, pn, ACT_NONE);
    float* u = (float*)A.take(512L * 16384 * 4);
    conv_run(st, A, p2b, 256, 130, 130, e.c4w, e.c4b, 512, 3, 3, 1, 0, u, 128, 128, ACT_TANH);

    // per-class masked average of u over seg128
    float* area = (float*)A.take(19 * 4);
    float* sums = (float*)A.take(19 * 512 * 4);
    k_fill_f32<<<1, 32, 0, st>>>(area, 0.f, 19);
    k_fill_f32<<<gs(19 * 512), 256, 0, st>>>(sums, 0.f, 19 * 512);
    k_area<<<gs(16384), 256, 0, st>>>(lab256, area, 16384);
    k_class_sum<<<gs(512L * 16384), 256, 0, st>>>(u, lab256, sums, 16384, 512L * 16384);
    k_style_final<<<gs(19 * 512), 256, 0, st>>>(sums, area, style, 19 * 512);
  }
}

}  // namespace

extern "C" void kernel_launch(void* const* d_in, const int* in_sizes, int n_in,
                              void* d_out, int out_size, void* d_ws, size_t ws_size,
                              hipStream_t stream) {
  (void)in_sizes; (void)n_in; (void)out_size;
  const float* img = (const float*)d_in[0];
  const float* seg = (const float*)d_in[1];

  // ---- walk flattened params (setup_inputs dict-insertion order) ----
  int cur = 2;
  auto nx = [&]() { return (const float*)d_in[cur++]; };
  auto parse_detail = [&](bool stl) {
    DetailP d{};
    d.noise_var = nx(); d.bg = nx(); d.bb = nx();
    d.sp_w = nx(); d.sp_b = nx();
    d.g_w = nx(); d.g_b = nx(); d.b_w = nx(); d.b_b = nx();
    if (stl) { d.st_w = nx(); d.st_b = nx(); d.cg_w = nx(); d.cg_b = nx(); d.cb_w = nx(); d.cb_b = nx(); }
    return d;
  };
  auto parse_res = [&](bool stl) {
    ResP r{};
    r.d0 = parse_detail(stl); r.d1 = parse_detail(stl); r.ds = parse_detail(stl);
    r.c0_w = nx(); r.c0_b = nx(); r.c1_w = nx(); r.c1_b = nx(); r.cs_w = nx();
    return r;
  };
  EncP e{};
  e.c1w = nx(); e.c1b = nx(); e.c2w = nx(); e.c2b = nx();
  e.c3w = nx(); e.c3b = nx(); e.ctw = nx(); e.ctb = nx();
  e.c4w = nx(); e.c4b = nx();
  const float* head_w = nx();
  const float* head_b = nx();
  ResP R[7];
  for (int i = 0; i < 7; ++i) R[i] = parse_res(i < 6);
  const float* out_w = nx();
  const float* out_b = nx();

  // ---- workspace arena (deterministic layout, graph-capture safe) ----
  Arena A{(char*)d_ws, 0, ws_size};
  int* lab256 = (int*)A.take(65536 * 4);
  float* style = (float*)A.take(19 * 512 * 4);
  float* hb0 = (float*)A.take((size_t)64 * 1024 * 1024 * 4);  // max h: 64 x 1024^2
  float* hb1 = (float*)A.take((size_t)64 * 1024 * 1024 * 4);
  size_t base_mark = A.off;

  k_labels<<<gs(65536), 256, 0, stream>>>(seg, lab256, 65536);

  encoder_run(stream, A, e, img, lab256, style);
  A.off = base_mark;

  // head: conv(nearest(seg,8,8)) -> (1024,8,8)
  {
    bf16* s8 = (bf16*)A.take((size_t)19 * 64 * sizeof(bf16));
    k_segmap<<<gs(19L * 64), 256, 0, stream>>>(lab256, s8, 8, 19L * 64);
    conv_run(stream, A, s8, 19, 8, 8, head_w, head_b, 1024, 3, 3, 1, 1, hb0, 8, 8, ACT_NONE);
    A.off = base_mark;
  }

  float* h = hb0;
  float* hn = hb1;
  int S = 8;
  const int cins[7]  = {1024, 1024, 1024, 1024, 512, 256, 128};
  const int couts[7] = {1024, 1024, 1024, 512, 256, 128, 64};
  for (int i = 0; i < 7; ++i) {
    float* r = resblk_run(stream, A, R[i], cins[i], couts[i], i < 6, h, S, lab256, style);
    long on = (long)couts[i] * 4 * S * S;
    k_upsample<<<gs(on), 256, 0, stream>>>(r, couts[i], S, S, hn, 2 * S, 2 * S, on);
    A.off = base_mark;
    float* t = h; h = hn; hn = t;
    S *= 2;
  }

  // final: tanh(conv(lrelu(h))) -> (3,1024,1024) into d_out
  {
    long n = 64L * 1024 * 1024;
    bf16* xb = (bf16*)A.take((size_t)n * sizeof(bf16));
    k_f32_to_bf16_act<<<gs(n), 256, 0, stream>>>(h, xb, n, ACT_LRELU);
    conv_run(stream, A, xb, 64, 1024, 1024, out_w, out_b, 3, 3, 3, 1, 1,
             (float*)d_out, 1024, 1024, ACT_TANH);
  }
}
